// Attention_54520314856023
// MI455X (gfx1250) — matrix operand order repaired and checked by reading, not compiled, not run on hardware
//
#include <hip/hip_runtime.h>
#include <hip/hip_bf16.h>

#define B   8
#define SQ  2048
#define SKV 2048
#define DIN 1024
#define H   512

typedef __attribute__((ext_vector_type(16))) __bf16 v16bf;
typedef __attribute__((ext_vector_type(8)))  __bf16 v8bf;
typedef __attribute__((ext_vector_type(4)))  __bf16 v4bf;
typedef __attribute__((ext_vector_type(8)))  float  v8f;
typedef __attribute__((ext_vector_type(4)))  float  v4f;
typedef __attribute__((ext_vector_type(4)))  int    v4i;

#define WMMA_BF16(a, b, c) \
  __builtin_amdgcn_wmma_f32_16x16x32_bf16(false, (a), false, (b), (short)0, (c), false, false)

// ---------------------------------------------------------------------------
// Async global->LDS staging (CDNA5 GLOBAL_LOAD_ASYNC_TO_LDS_B128, ASYNCcnt)
// with a synchronous fallback if the builtin is unavailable.
// Builtin signature (from hipcc diagnostic): (v4i AS1*, v4i AS3*, imm, imm)
// ---------------------------------------------------------------------------
#if __has_builtin(__builtin_amdgcn_global_load_async_to_lds_b128)
#define HAVE_ASYNC_LDS 1
#else
#define HAVE_ASYNC_LDS 0
#endif

static __device__ __forceinline__ void async_copy16(const __bf16* g, __bf16* l) {
#if HAVE_ASYNC_LDS
  __bf16* gm = const_cast<__bf16*>(g);
  __builtin_amdgcn_global_load_async_to_lds_b128(
      (__attribute__((address_space(1))) v4i*)gm,
      (__attribute__((address_space(3))) v4i*)l, 0, 0);
#else
  *(v8bf*)l = *(const v8bf*)g;
#endif
}

static __device__ __forceinline__ void wait_async_lds() {
#if HAVE_ASYNC_LDS
#if __has_builtin(__builtin_amdgcn_s_wait_asynccnt)
  __builtin_amdgcn_s_wait_asynccnt(0);
#else
  asm volatile("s_wait_asynccnt 0x0" ::: "memory");
#endif
#endif
}

// ---------------------------------------------------------------------------
// A-fragment (16x32 bf16, MxK) from an LDS-resident row-major tile.
//   lanes 0-15  (M = lane):    halfs K 0..7  and K 8..15
//   lanes 16-31 (M = lane-16): halfs K 16..23 and K 24..31
// ---------------------------------------------------------------------------
#define LDS_A_FRAG(frag, ldsArr, k0)                                           \
  do {                                                                         \
    const int _lane = threadIdx.x & 31;                                        \
    const int _row = _lane & 15;                                               \
    const int _kb = (_lane >> 4) * 16;                                         \
    v8bf _lo = *(const v8bf*)&ldsArr[_row][(k0) + _kb];                        \
    v8bf _hi = *(const v8bf*)&ldsArr[_row][(k0) + _kb + 8];                    \
    _Pragma("unroll") for (int _i = 0; _i < 8; ++_i) {                         \
      frag[_i] = _lo[_i];                                                      \
      frag[_i + 8] = _hi[_i];                                                  \
    }                                                                          \
  } while (0)

// A-fragment straight from global (row-major, leading dim ld)
static __device__ __forceinline__ v16bf load_a_frag(const __bf16* __restrict__ base,
                                                    int ld, int k0) {
  const int lane = threadIdx.x & 31;
  const int row  = lane & 15;
  const int kb   = (lane >> 4) * 8;
  const __bf16* p = base + (size_t)row * ld + k0 + kb;
  v8bf lo = *(const v8bf*)p;
  v8bf hi = *(const v8bf*)(p + 16);
  v16bf r;
#pragma unroll
  for (int i = 0; i < 8; ++i) { r[i] = lo[i]; r[i + 8] = hi[i]; }
  return r;
}

// B-fragment (32x16 bf16, KxN): per lane 16 contiguous K-values of column
// n = n0 + (lane&15), K starting at k0 + (lane>=16 ? 16 : 0). Source stored
// "transposed" T[n][k] row-major => one contiguous 32-byte load per lane.
static __device__ __forceinline__ v16bf load_b_frag(const __bf16* __restrict__ baseT,
                                                    int ld, int n0, int k0) {
  const int lane = threadIdx.x & 31;
  const int n  = n0 + (lane & 15);
  const int kb = (lane >> 4) * 16;
  return *(const v16bf*)(baseT + (size_t)n * ld + k0 + kb);
}

// ---------------------------------------------------------------------------
// fp32 -> bf16 elementwise (vector x4)
// ---------------------------------------------------------------------------
__global__ void cvt_f32_bf16(const float* __restrict__ in, __bf16* __restrict__ out, int n) {
  int i = (blockIdx.x * blockDim.x + threadIdx.x) * 4;
  if (i + 3 < n) {
    v4f v = *(const v4f*)(in + i);
    v4bf o;
#pragma unroll
    for (int k = 0; k < 4; ++k) o[k] = (__bf16)v[k];
    *(v4bf*)(out + i) = o;
  }
}

// W[din][h] (f32) -> Wt[h][din] (bf16)
__global__ void transpose_w(const float* __restrict__ w, __bf16* __restrict__ wt) {
  int idx = blockIdx.x * blockDim.x + threadIdx.x;   // over H*DIN
  int h = idx / DIN, d = idx % DIN;
  wt[idx] = (__bf16)w[(size_t)d * H + h];
}

// ---------------------------------------------------------------------------
// Fused projection: one block = one 16-row m-tile of states (staged to LDS,
// 32KB, async). Waves 0-3 produce K [b][kv][h], waves 4-7 produce Vt [b][h][kv].
// Each wave owns a 16x128 output strip: 8 accumulators, A-frag reused 8x.
// ---------------------------------------------------------------------------
__global__ void proj_kernel(const __bf16* __restrict__ sbf,
                            const __bf16* __restrict__ wkt, const __bf16* __restrict__ wvt,
                            const float* __restrict__ bk, const float* __restrict__ bv,
                            __bf16* __restrict__ kout, __bf16* __restrict__ vtout) {
  __shared__ __bf16 ldsS[16][DIN + 8];
  const int wave = threadIdx.x >> 5;
  const int lane = threadIdx.x & 31;
  const int mtile = blockIdx.x;                    // 0..1023 (B*SKV/16)
  const __bf16* sg = sbf + (size_t)(mtile * 16) * DIN;

  // stage 16x1024 bf16 (32KB): 2048 x 16B, 8 per thread
#pragma unroll
  for (int i = 0; i < 8; ++i) {
    int t = threadIdx.x + i * 256;
    int row = t >> 7;                              // 128 chunks per row
    int col = (t & 127) * 8;
    async_copy16(sg + (size_t)row * DIN + col, &ldsS[row][col]);
  }
  wait_async_lds();
  __syncthreads();

  const int isV = wave >> 2;
  const __bf16* wt  = isV ? wvt : wkt;
  const float* bias = isV ? bv : bk;
  const int nbase = (wave & 3) * 128;

  v8f acc[8] = {};
  for (int k0 = 0; k0 < DIN; k0 += 32) {
    v16bf a;
    LDS_A_FRAG(a, ldsS, k0);
#pragma unroll
    for (int j = 0; j < 8; ++j) {
      v16bf bf = load_b_frag(wt, DIN, nbase + j * 16, k0);
      acc[j] = WMMA_BF16(a, bf, acc[j]);
    }
  }

  const int nlo = lane & 15;
  const int mo  = (lane >> 4) * 8;
#pragma unroll
  for (int j = 0; j < 8; ++j) {
    const int n = nbase + j * 16 + nlo;
    const float bn = bias[n];
    if (isV) {                                     // Vt [b][h][kv], packed 16B store
      const int mbase = mtile * 16 + mo;
      const int bb = mbase >> 11;
      const int kv = mbase & (SKV - 1);
      v8bf pk;
#pragma unroll
      for (int r = 0; r < 8; ++r) pk[r] = (__bf16)(acc[j][r] + bn);
      *(v8bf*)(vtout + ((size_t)bb * H + n) * SKV + kv) = pk;
    } else {                                       // K [b][kv][h], b16 coalesced
#pragma unroll
      for (int r = 0; r < 8; ++r) {
        int m = mtile * 16 + r + mo;
        kout[(size_t)m * H + n] = (__bf16)(acc[j][r] + bn);
      }
    }
  }
}

// ---------------------------------------------------------------------------
// Scores + masked exp + row-sum. One block = one (b, 16-row q-tile). Q tile
// (16x512 bf16, 16KB) staged once to LDS and shared by all 8 waves; each wave
// covers 256 kv columns, 4 tiles at a time (A-frag reused 4x, 4 indep WMMA).
// p = exp(scale*s*mask) stored unnormalized (bf16); rowsums reduced cross-lane
// then cross-wave; 1/rowsum folded into the P@V kernel.
// ---------------------------------------------------------------------------
__global__ void scores_kernel(const __bf16* __restrict__ qbf, const __bf16* __restrict__ kbf,
                              const float* __restrict__ mask, __bf16* __restrict__ pu,
                              float* __restrict__ rowsum) {
  __shared__ __bf16 ldsQ[16][H + 8];
  __shared__ float red[8][16];
  const int wave = threadIdx.x >> 5;
  const int lane = threadIdx.x & 31;
  const int b  = blockIdx.x >> 7;
  const int q0 = (blockIdx.x & 127) * 16;
  const float scale = 0.044194173824159216f;       // 1/sqrt(512)

  const __bf16* qg = qbf + ((size_t)b * SQ + q0) * H;
  const __bf16* kb = kbf + (size_t)b * SKV * H;    // [kv][h] == T[n][k]

  // stage 16x512 bf16 (16KB): 1024 x 16B, 4 per thread
#pragma unroll
  for (int i = 0; i < 4; ++i) {
    int t = threadIdx.x + i * 256;
    int row = t >> 6;                              // 64 chunks per row
    int col = (t & 63) * 8;
    async_copy16(qg + (size_t)row * H + col, &ldsQ[row][col]);
  }
  wait_async_lds();
  __syncthreads();

  const int nlo = lane & 15;
  const int mo  = (lane >> 4) * 8;
  float rs[8];
#pragma unroll
  for (int r = 0; r < 8; ++r) rs[r] = 0.f;

  for (int tt = 0; tt < 4; ++tt) {                 // 16 kv tiles, groups of 4
    const int kvbase = wave * 256 + tt * 64;
    v8f c[4] = {};
    for (int k0 = 0; k0 < H; k0 += 32) {
      v16bf a;
      LDS_A_FRAG(a, ldsQ, k0);
#pragma unroll
      for (int j = 0; j < 4; ++j) {
        v16bf bf = load_b_frag(kb, H, kvbase + j * 16, k0);
        c[j] = WMMA_BF16(a, bf, c[j]);
      }
    }
#pragma unroll
    for (int j = 0; j < 4; ++j) {
      const int n = kvbase + j * 16 + nlo;
#pragma unroll
      for (int r = 0; r < 8; ++r) {
        const int m = q0 + r + mo;
        const size_t off = ((size_t)b * SQ + m) * SKV + n;
        float p = __expf(c[j][r] * scale * mask[off]);
        rs[r] += p;
        pu[off] = (__bf16)p;
      }
    }
  }

  // reduce across the 16 lanes holding one row (xor stays within the half)
#pragma unroll
  for (int r = 0; r < 8; ++r) {
    float v = rs[r];
    v += __shfl_xor(v, 1, 32);
    v += __shfl_xor(v, 2, 32);
    v += __shfl_xor(v, 4, 32);
    v += __shfl_xor(v, 8, 32);
    rs[r] = v;
  }
  if (nlo == 0) {                                  // lanes 0 / 16
#pragma unroll
    for (int r = 0; r < 8; ++r) red[wave][r + mo] = rs[r];
  }
  __syncthreads();
  if (threadIdx.x < 16) {
    float tot = 0.f;
#pragma unroll
    for (int w = 0; w < 8; ++w) tot += red[w][threadIdx.x];
    rowsum[(size_t)b * SQ + q0 + threadIdx.x] = tot;
  }
}

// ---------------------------------------------------------------------------
// out = (Pu @ V) / rowsum. One block = one (b, 16-row q-tile) x full H.
// P is staged to LDS in 16x512 chunks (16KB, async) shared by all 8 waves;
// each wave owns a 16x64 output strip (4 accumulators, A-frag reused 4x).
// ---------------------------------------------------------------------------
__global__ void out_kernel(const __bf16* __restrict__ pu, const __bf16* __restrict__ vt,
                           const float* __restrict__ rowsum, float* __restrict__ out) {
  __shared__ __bf16 ldsP[16][512 + 8];
  const int wave = threadIdx.x >> 5;
  const int lane = threadIdx.x & 31;
  const int b  = blockIdx.x >> 7;
  const int q0 = (blockIdx.x & 127) * 16;

  const __bf16* pb = pu + ((size_t)b * SQ + q0) * SKV;
  const __bf16* vb = vt + (size_t)b * H * SKV;

  v8f acc[4] = {};
  for (int kc = 0; kc < SKV; kc += 512) {
    __syncthreads();                               // protect previous chunk reads
#pragma unroll
    for (int i = 0; i < 4; ++i) {
      int t = threadIdx.x + i * 256;
      int row = t >> 6;
      int col = (t & 63) * 8;
      async_copy16(pb + (size_t)row * SKV + kc + col, &ldsP[row][col]);
    }
    wait_async_lds();
    __syncthreads();

    for (int k0 = 0; k0 < 512; k0 += 32) {
      v16bf a;
      LDS_A_FRAG(a, ldsP, k0);
#pragma unroll
      for (int j = 0; j < 4; ++j) {
        const int h0 = (wave * 4 + j) * 16;
        v16bf bf = load_b_frag(vb, SKV, h0, kc + k0);
        acc[j] = WMMA_BF16(a, bf, acc[j]);
      }
    }
  }

  const int nlo = lane & 15;
  const int mo  = (lane >> 4) * 8;
#pragma unroll
  for (int j = 0; j < 4; ++j) {
    const int n = (wave * 4 + j) * 16 + nlo;
#pragma unroll
    for (int r = 0; r < 8; ++r) {
      const int m = q0 + r + mo;
      const float inv = 1.0f / rowsum[(size_t)b * SQ + m];
      out[((size_t)b * SQ + m) * H + n] = acc[j][r] * inv;
    }
  }
}

// ---------------------------------------------------------------------------
extern "C" void kernel_launch(void* const* d_in, const int* in_sizes, int n_in,
                              void* d_out, int out_size, void* d_ws, size_t ws_size,
                              hipStream_t stream) {
  const float* query  = (const float*)d_in[0];
  const float* states = (const float*)d_in[1];
  const float* mask   = (const float*)d_in[2];
  const float* Wk     = (const float*)d_in[3];
  const float* bk     = (const float*)d_in[4];
  const float* Wv     = (const float*)d_in[5];
  const float* bv     = (const float*)d_in[6];
  float* out = (float*)d_out;

  // workspace carve (~153 MB): bf16 intermediates + f32 rowsums
  char* w = (char*)d_ws;
  __bf16* qbf = (__bf16*)w; w += (size_t)B * SQ  * H   * 2;
  __bf16* sbf = (__bf16*)w; w += (size_t)B * SKV * DIN * 2;
  __bf16* wkt = (__bf16*)w; w += (size_t)H * DIN * 2;
  __bf16* wvt = (__bf16*)w; w += (size_t)H * DIN * 2;
  __bf16* kbf = (__bf16*)w; w += (size_t)B * SKV * H * 2;   // K  [b][kv][h]
  __bf16* vtb = (__bf16*)w; w += (size_t)B * H * SKV * 2;   // Vt [b][h][kv]
  __bf16* pu  = (__bf16*)w; w += (size_t)B * SQ * SKV * 2;  // unnormalized P
  float*  rowsum = (float*)w;

  cvt_f32_bf16<<<(B * SQ  * H   / 4 + 255) / 256, 256, 0, stream>>>(query,  qbf, B * SQ  * H);
  cvt_f32_bf16<<<(B * SKV * DIN / 4 + 255) / 256, 256, 0, stream>>>(states, sbf, B * SKV * DIN);
  transpose_w<<<H * DIN / 256, 256, 0, stream>>>(Wk, wkt);
  transpose_w<<<H * DIN / 256, 256, 0, stream>>>(Wv, wvt);

  proj_kernel<<<B * SKV / 16, 256, 0, stream>>>(sbf, wkt, wvt, bk, bv, kbf, vtb);
  scores_kernel<<<B * SQ / 16, 256, 0, stream>>>(qbf, kbf, mask, pu, rowsum);
  out_kernel<<<B * SQ / 16, 256, 0, stream>>>(pu, vtb, rowsum, out);
}
